// Music_transformer_34565896798700
// MI455X (gfx1250) — matrix operand order repaired and checked by reading, not compiled, not run on hardware
//
#include <hip/hip_runtime.h>

// ---------------------------------------------------------------------------
// Music transformer block (GTrXL-style) for MI455X / gfx1250.
// All matmuls via v_wmma_f32_16x16x32_bf16 (bf16 in, f32 accumulate).
// GEMM: 256 thr (8 wave32), block tile 128x64, wave tile 32x32 (4 WMMA/step),
// template-specialized on A-precision / B-layout / epilogue flags.
// ---------------------------------------------------------------------------

#define BB 8
#define SS 512
#define MMEM 512
#define DD 1024
#define HH 16
#define FF 1024
#define DHH 64

typedef __attribute__((ext_vector_type(16))) __bf16 v16bf;
typedef __attribute__((ext_vector_type(8)))  float v8f;

union FragU { unsigned int u[8]; v16bf v; };

enum { GF_ACC = 1, GF_GELU = 2, GF_SHIFT = 4 };

__device__ __forceinline__ unsigned short f2bf(float f) {
  unsigned int u = __float_as_uint(f);
  u += 0x7FFFu + ((u >> 16) & 1u);   // round-to-nearest-even
  return (unsigned short)(u >> 16);
}
__device__ __forceinline__ float geluf(float x) {
  return 0.5f * x * (1.0f + erff(x * 0.7071067811865475f));
}
__device__ __forceinline__ float sigmoidf_(float x) { return 1.0f / (1.0f + expf(-x)); }

// ---------------------------------------------------------------------------
// Generic batched GEMM: C[z] (M x N) = A[z] (M x K) * B[z]
//  AISF32: A operand is f32 (converted to bf16 during LDS staging)
//  BMODE : 0 -> B is (N x K) contiguous along K ("pre-transposed weight")
//          1 -> B is (K x N) contiguous along N (e.g. attn @ v)
//  FLAGS : GF_ACC   C += A*B
//          GF_GELU  C = gelu(A*B + bias)
//          GF_SHIFT relative-skew scatter C[m][n+m-shiftBase] += (A*B)[m][n]
// z = b*Hh + h with separate b/h strides so per-(batch, head) slices of
// packed (B,*,H,DH) tensors address naturally.
// ---------------------------------------------------------------------------
template<int AISF32, int BMODE, int FLAGS>
__global__ __launch_bounds__(256)
void gemm_bf16_wmma(const void* __restrict__ Aptr, long lda, long sAb, long sAh,
                    const unsigned short* __restrict__ Bm,
                    long ldb, long sBb, long sBh,
                    float* __restrict__ C, long ldc, long sCb, long sCh,
                    unsigned short* __restrict__ Cbf, long ldcb, long sCbb, long sCbh,
                    const float* __restrict__ bias,
                    int Kdim, int Ncols, int Hh, int shiftBase)
{
  __shared__ unsigned short lA[128 * 32];     // A tile, m-major
  __shared__ unsigned short lB[64 * 32];      // B tile stored n-major (transposed)

  const int z  = blockIdx.z;
  const int bb = z / Hh, hh = z % Hh;
  const long aoff  = (long)bb * sAb  + (long)hh * sAh;
  const long boff  = (long)bb * sBb  + (long)hh * sBh;
  const long coff  = (long)bb * sCb  + (long)hh * sCh;
  const long cboff = (long)bb * sCbb + (long)hh * sCbh;

  const int m0 = blockIdx.y * 128;
  const int n0 = blockIdx.x * 64;
  const int tid = threadIdx.x;
  const int lane = tid & 31;
  const int w = tid >> 5;
  const int r0 = (w & 3) * 32;                // wave row block (32 rows)
  const int c0 = (w >> 2) * 32;               // wave col block (32 cols)

  const unsigned short* A16 = (const unsigned short*)Aptr;
  const float*          A32 = (const float*)Aptr;

  v8f acc[2][2];
#pragma unroll
  for (int mi = 0; mi < 2; ++mi)
#pragma unroll
    for (int ni = 0; ni < 2; ++ni)
#pragma unroll
      for (int i = 0; i < 8; ++i) acc[mi][ni][i] = 0.f;

  const int srowA = tid >> 1;                 // 0..127
  const int scolA = (tid & 1) * 16;           // 0,16
  const int srowB = tid >> 2;                 // 0..63
  const int scolB = (tid & 3) * 8;            // 0,8,16,24

  for (int k0 = 0; k0 < Kdim; k0 += 32) {
    // --- stage A (128 x 32) ---
    {
      long base = aoff + (long)(m0 + srowA) * lda + (k0 + scolA);
      if (AISF32) {
#pragma unroll
        for (int j = 0; j < 16; ++j) lA[srowA * 32 + scolA + j] = f2bf(A32[base + j]);
      } else {
        const unsigned int* p = (const unsigned int*)(A16 + base);
        unsigned int* d = (unsigned int*)&lA[srowA * 32 + scolA];
#pragma unroll
        for (int j = 0; j < 8; ++j) d[j] = p[j];
      }
    }
    // --- stage B (64 x 32), write n-major into lB ---
    if (BMODE == 0) {
      long base = boff + (long)(n0 + srowB) * ldb + (k0 + scolB);
      const unsigned int* p = (const unsigned int*)(Bm + base);
      unsigned int* d = (unsigned int*)&lB[srowB * 32 + scolB];
#pragma unroll
      for (int j = 0; j < 4; ++j) d[j] = p[j];
    } else {
#pragma unroll
      for (int j = 0; j < 8; ++j)
        lB[srowB * 32 + scolB + j] = Bm[boff + (long)(k0 + scolB + j) * ldb + (n0 + srowB)];
    }
    __syncthreads();

    // --- build fragments (ISA 7.12.2 16-bit layouts) ---
    FragU fa[2], fb[2];
    const int hi = lane >> 4;
#pragma unroll
    for (int mi = 0; mi < 2; ++mi) {
      const int mrow = r0 + mi * 16 + (lane & 15);
#pragma unroll
      for (int j = 0; j < 8; ++j) {
        int kk = (j < 4) ? (8 * hi + 2 * j) : (16 + 8 * hi + 2 * (j - 4));
        fa[mi].u[j] = *(const unsigned int*)&lA[mrow * 32 + kk];
      }
    }
    const int kb = hi * 16;
#pragma unroll
    for (int ni = 0; ni < 2; ++ni) {
      const int nc = c0 + ni * 16 + (lane & 15);
#pragma unroll
      for (int j = 0; j < 8; ++j)
        fb[ni].u[j] = *(const unsigned int*)&lB[nc * 32 + ((j < 4) ? (8 * hi + 2 * j) : (16 + 8 * hi + 2 * (j - 4)))];
    }
#pragma unroll
    for (int mi = 0; mi < 2; ++mi)
#pragma unroll
      for (int ni = 0; ni < 2; ++ni)
        acc[mi][ni] = __builtin_amdgcn_wmma_f32_16x16x32_bf16(
            false, fa[mi].v, false, fb[ni].v, (short)0, acc[mi][ni], false, false);
    __syncthreads();
  }

  // --- epilogue ---
#pragma unroll
  for (int mi = 0; mi < 2; ++mi) {
    const int mb = m0 + r0 + mi * 16 + (lane >> 4) * 8;
#pragma unroll
    for (int ni = 0; ni < 2; ++ni) {
      const int nn = n0 + c0 + ni * 16 + (lane & 15);
#pragma unroll
      for (int i = 0; i < 8; ++i) {
        const int m = mb + i;
        float val = acc[mi][ni][i];
        if (FLAGS & GF_SHIFT) {
          // relative-position skew: target column j = n + m - (S-1); unique per
          // element across the whole grid, so plain += is race-free.
          int jj = nn + m - shiftBase;
          if (jj >= 0 && jj < Ncols) C[coff + (long)m * ldc + jj] += val;
        } else if (FLAGS & GF_ACC) {
          C[coff + (long)m * ldc + nn] += val;
        } else {
          if (bias) val += bias[nn];
          if (FLAGS & GF_GELU) val = geluf(val);
          if (C)   C[coff + (long)m * ldc + nn] = val;
          if (Cbf) Cbf[cboff + (long)m * ldcb + nn] = f2bf(val);
        }
      }
    }
  }
}

// ---------------------------------------------------------------------------
// Weight transpose + f32->bf16: Wt[n*K+k] = bf16(W[k*N+n])
// ---------------------------------------------------------------------------
__global__ __launch_bounds__(256)
void transpose_to_bf16_kernel(const float* __restrict__ W, unsigned short* __restrict__ Wt,
                              int K, int N)
{
  __shared__ float tile[32][33];
  const int bx = blockIdx.x * 32;   // n
  const int by = blockIdx.y * 32;   // k
  const int tx = threadIdx.x & 31;
  const int ty = threadIdx.x >> 5;  // 0..7
  for (int i = ty; i < 32; i += 8) tile[i][tx] = W[(long)(by + i) * N + (bx + tx)];
  __syncthreads();
  for (int i = ty; i < 32; i += 8) Wt[(long)(bx + i) * K + (by + tx)] = f2bf(tile[tx][i]);
}

__global__ void f32_to_bf16_kernel(const float* __restrict__ in,
                                   unsigned short* __restrict__ out, long n)
{
  long i = (long)blockIdx.x * 256 + threadIdx.x;
  if (i < n) out[i] = f2bf(in[i]);
}

// ---------------------------------------------------------------------------
// LayerNorm over concat(mem, x) -> bf16 xt (B,F,D); also writes xq rows.
// ---------------------------------------------------------------------------
__global__ __launch_bounds__(256)
void ln_concat_kernel(const float* __restrict__ mem, const float* __restrict__ x,
                      const float* __restrict__ g, const float* __restrict__ bta,
                      unsigned short* __restrict__ xt, unsigned short* __restrict__ xq)
{
  __shared__ float red[256];
  const int row = blockIdx.x;
  const int b = row / FF, f = row % FF;
  const float* src = (f < MMEM) ? (mem + ((long)b * MMEM + f) * DD)
                                : (x + ((long)b * SS + (f - MMEM)) * DD);
  const int tid = threadIdx.x;
  float s = 0.f, s2 = 0.f;
  for (int i = tid; i < DD; i += 256) { float v = src[i]; s += v; s2 += v * v; }
  red[tid] = s; __syncthreads();
  for (int o = 128; o > 0; o >>= 1) { if (tid < o) red[tid] += red[tid + o]; __syncthreads(); }
  const float mu = red[0] / (float)DD; __syncthreads();
  red[tid] = s2; __syncthreads();
  for (int o = 128; o > 0; o >>= 1) { if (tid < o) red[tid] += red[tid + o]; __syncthreads(); }
  const float var = red[0] / (float)DD - mu * mu;
  const float rstd = rsqrtf(var + 1e-5f);
  unsigned short* dt = xt + (long)row * DD;
  unsigned short* dq = (f >= MMEM) ? (xq + ((long)b * SS + (f - MMEM)) * DD) : nullptr;
  for (int i = tid; i < DD; i += 256) {
    float v = (src[i] - mu) * rstd * g[i] + bta[i];
    unsigned short h = f2bf(v);
    dt[i] = h;
    if (dq) dq[i] = h;
  }
}

__global__ __launch_bounds__(256)
void ln_simple_kernel(const float* __restrict__ in, const float* __restrict__ g,
                      const float* __restrict__ bta, unsigned short* __restrict__ outbf)
{
  __shared__ float red[256];
  const long row = blockIdx.x;
  const float* src = in + row * DD;
  const int tid = threadIdx.x;
  float s = 0.f, s2 = 0.f;
  for (int i = tid; i < DD; i += 256) { float v = src[i]; s += v; s2 += v * v; }
  red[tid] = s; __syncthreads();
  for (int o = 128; o > 0; o >>= 1) { if (tid < o) red[tid] += red[tid + o]; __syncthreads(); }
  const float mu = red[0] / (float)DD; __syncthreads();
  red[tid] = s2; __syncthreads();
  for (int o = 128; o > 0; o >>= 1) { if (tid < o) red[tid] += red[tid + o]; __syncthreads(); }
  const float var = red[0] / (float)DD - mu * mu;
  const float rstd = rsqrtf(var + 1e-5f);
  for (int i = tid; i < DD; i += 256)
    outbf[row * DD + i] = f2bf((src[i] - mu) * rstd * g[i] + bta[i]);
}

// ---------------------------------------------------------------------------
// Softmax over score rows (length F) with fused scale + causal mask.
// Row max-prob == 1/sum -> atomicMax per (b,h) for attention_loss.
// ---------------------------------------------------------------------------
__global__ __launch_bounds__(256)
void softmax_kernel(float* __restrict__ attn, float* __restrict__ lossmax)
{
  __shared__ float red[256];
  const long row = blockIdx.x;            // 0 .. B*H*S-1
  const int  s   = (int)(row % SS);
  const int  bh  = (int)(row / SS);
  float* p = attn + row * (long)FF;
  const int tid = threadIdx.x;
  const float scale = 0.03125f;           // 1/sqrt(D) = 1/32
  float lv[FF / 256];
#pragma unroll
  for (int jj = 0; jj < FF / 256; ++jj) {
    int j = tid + jj * 256;
    lv[jj] = p[j] * scale + ((j > s + MMEM) ? -1.0e10f : 0.0f);
  }
  float mx = -3.4e38f;
#pragma unroll
  for (int jj = 0; jj < FF / 256; ++jj) mx = fmaxf(mx, lv[jj]);
  red[tid] = mx; __syncthreads();
  for (int o = 128; o > 0; o >>= 1) { if (tid < o) red[tid] = fmaxf(red[tid], red[tid + o]); __syncthreads(); }
  mx = red[0]; __syncthreads();
  float sum = 0.f;
#pragma unroll
  for (int jj = 0; jj < FF / 256; ++jj) { lv[jj] = expf(lv[jj] - mx); sum += lv[jj]; }
  red[tid] = sum; __syncthreads();
  for (int o = 128; o > 0; o >>= 1) { if (tid < o) red[tid] += red[tid + o]; __syncthreads(); }
  const float inv = 1.0f / red[0];
#pragma unroll
  for (int jj = 0; jj < FF / 256; ++jj) p[tid + jj * 256] = lv[jj] * inv;
  if (tid == 0)
    atomicMax((unsigned int*)&lossmax[bh], __float_as_uint(inv));  // max prob = 1/sum
}

__global__ void zero128_kernel(float* p) { p[threadIdx.x] = 0.f; }

__global__ void loss_finalize_kernel(const float* __restrict__ lossmax, float* __restrict__ out)
{
  __shared__ float red[128];
  const int t = threadIdx.x;
  red[t] = lossmax[t]; __syncthreads();
  for (int o = 64; o > 0; o >>= 1) { if (t < o) red[t] += red[t + o]; __syncthreads(); }
  if (t == 0) out[0] = red[0] / (float)(BB * HH);
}

// GRU elementwise stages
__global__ void gru_rx_kernel(const float* __restrict__ t1, const float* __restrict__ xf,
                              unsigned short* __restrict__ rx_bf, long n)
{
  long i = (long)blockIdx.x * 256 + threadIdx.x;
  if (i < n) rx_bf[i] = f2bf(sigmoidf_(t1[i]) * xf[i]);
}

__global__ void gru_out_kernel(const float* __restrict__ t2, const float* __restrict__ t3,
                               const float* __restrict__ xf, const float* __restrict__ bg,
                               float* __restrict__ outf, unsigned short* __restrict__ outbf, long n)
{
  long i = (long)blockIdx.x * 256 + threadIdx.x;
  if (i < n) {
    float zz = sigmoidf_(t2[i] - bg[i % DD]);
    float gg = (1.0f - zz) * xf[i] + zz * tanhf(t3[i]);
    if (outf)  outf[i] = gg;
    if (outbf) outbf[i] = f2bf(gg);
  }
}

// ---------------------------------------------------------------------------
// Host side
// ---------------------------------------------------------------------------
static inline void launch_gemm(hipStream_t st,
    const void* A, int aIsF32, long lda, long sAb, long sAh,
    const unsigned short* Bm, long ldb, long sBb, long sBh, int bmode,
    float* C, long ldc, long sCb, long sCh,
    unsigned short* Cbf, long ldcb, long sCbb, long sCbh,
    const float* bias, int M, int N, int K, int Hh, int Z, int flags, int shiftBase)
{
  dim3 grid((unsigned)(N / 64), (unsigned)(M / 128), (unsigned)Z);
  dim3 blk(256);
  (void)bmode;
  if (aIsF32) {
    gemm_bf16_wmma<1, 1, 0><<<grid, blk, 0, st>>>(A, lda, sAb, sAh, Bm, ldb, sBb, sBh,
        C, ldc, sCb, sCh, Cbf, ldcb, sCbb, sCbh, bias, K, N, Hh, shiftBase);
  } else if (flags == GF_ACC) {
    gemm_bf16_wmma<0, 0, GF_ACC><<<grid, blk, 0, st>>>(A, lda, sAb, sAh, Bm, ldb, sBb, sBh,
        C, ldc, sCb, sCh, Cbf, ldcb, sCbb, sCbh, bias, K, N, Hh, shiftBase);
  } else if (flags == GF_GELU) {
    gemm_bf16_wmma<0, 0, GF_GELU><<<grid, blk, 0, st>>>(A, lda, sAb, sAh, Bm, ldb, sBb, sBh,
        C, ldc, sCb, sCh, Cbf, ldcb, sCbb, sCbh, bias, K, N, Hh, shiftBase);
  } else if (flags == GF_SHIFT) {
    gemm_bf16_wmma<0, 0, GF_SHIFT><<<grid, blk, 0, st>>>(A, lda, sAb, sAh, Bm, ldb, sBb, sBh,
        C, ldc, sCb, sCh, Cbf, ldcb, sCbb, sCbh, bias, K, N, Hh, shiftBase);
  } else {
    gemm_bf16_wmma<0, 0, 0><<<grid, blk, 0, st>>>(A, lda, sAb, sAh, Bm, ldb, sBb, sBh,
        C, ldc, sCb, sCh, Cbf, ldcb, sCbb, sCbh, bias, K, N, Hh, shiftBase);
  }
}

extern "C" void kernel_launch(void* const* d_in, const int* in_sizes, int n_in,
                              void* d_out, int out_size, void* d_ws, size_t ws_size,
                              hipStream_t stream)
{
  (void)in_sizes; (void)n_in; (void)out_size; (void)ws_size;

  // setup_inputs dict order (nested dicts depth-first, insertion order):
  const float* x_in    = (const float*)d_in[0];
  const float* mem_in  = (const float*)d_in[1];
  /* mask d_in[2] computed analytically */
  const float* relenc  = (const float*)d_in[3];
  const float* u_param = (const float*)d_in[9];
  const float* v_param = (const float*)d_in[10];
  const float* ln1_g   = (const float*)d_in[11];
  const float* ln1_b   = (const float*)d_in[12];
  const float* ln2_g   = (const float*)d_in[13];
  const float* ln2_b   = (const float*)d_in[14];
  const float* lin1_b  = (const float*)d_in[16];
  const float* lin2_b  = (const float*)d_in[18];
  const float* bg1     = (const float*)d_in[25];
  const float* bg2     = (const float*)d_in[32];
  // D x D weights needing transpose+bf16: w_q w_k_e w_k_r w_v w_final lin1 lin2
  // g1{w_r u_r w_z u_z w_g u_g} g2{...}
  const int widx[19] = {4, 5, 6, 7, 8, 15, 17, 19, 20, 21, 22, 23, 24, 26, 27, 28, 29, 30, 31};

  const long BSD = (long)BB * SS * DD;   // 4194304
  const long BFD = (long)BB * FF * DD;   // 8388608
  const long ATT = (long)BB * HH * SS * FF;

  float* out_f = (float*)d_out;
  float* attn  = out_f + BSD;            // (B,H,S,F) f32 lives in d_out
  float* loss  = out_f + BSD + ATT;

  // -------- workspace bump allocator (256B aligned) --------
  char* wsp = (char*)d_ws;
  size_t off = 0;
  auto alloc = [&](size_t bytes) -> char* {
    char* p = wsp + off;
    off = (off + bytes + 255) & ~(size_t)255;
    return p;
  };
  unsigned short* wt[19];
  for (int i = 0; i < 19; ++i) wt[i] = (unsigned short*)alloc((size_t)DD * DD * 2);
  unsigned short* x_bf    = (unsigned short*)alloc(BSD * 2);
  unsigned short* re_bf   = (unsigned short*)alloc((size_t)FF * DD * 2);
  unsigned short* xt_bf   = (unsigned short*)alloc(BFD * 2);
  unsigned short* xq_bf   = (unsigned short*)alloc(BSD * 2);
  unsigned short* k_bf    = (unsigned short*)alloc(BFD * 2);
  unsigned short* v_bf    = (unsigned short*)alloc(BFD * 2);
  unsigned short* qu_bf   = (unsigned short*)alloc(BSD * 2);
  unsigned short* qv_bf   = (unsigned short*)alloc(BSD * 2);
  unsigned short* qrel_bf = (unsigned short*)alloc((size_t)FF * DD * 2);
  unsigned short* ao_bf   = (unsigned short*)alloc(BSD * 2);
  unsigned short* y_bf    = (unsigned short*)alloc(BSD * 2);
  unsigned short* rx_bf   = (unsigned short*)alloc(BSD * 2);
  unsigned short* g1_bf   = (unsigned short*)alloc(BSD * 2);
  unsigned short* h0_bf   = (unsigned short*)alloc(BSD * 2);
  unsigned short* h1_bf   = (unsigned short*)alloc(BSD * 2);
  unsigned short* h2_bf   = (unsigned short*)alloc(BSD * 2);
  float* t1     = (float*)alloc(BSD * 4);
  float* t2     = (float*)alloc(BSD * 4);
  float* t3     = (float*)alloc(BSD * 4);
  float* g1_f   = (float*)alloc(BSD * 4);
  float* lossmx = (float*)alloc(128 * 4);

  const long SD = (long)SS * DD, FD = (long)FF * DD, SF = (long)SS * FF, HSF = (long)HH * SF;

  // ---- weight prep (one-time per launch) ----
  for (int i = 0; i < 19; ++i)
    transpose_to_bf16_kernel<<<dim3(DD / 32, DD / 32), 256, 0, stream>>>(
        (const float*)d_in[widx[i]], wt[i], DD, DD);
  f32_to_bf16_kernel<<<(unsigned)((BSD + 255) / 256), 256, 0, stream>>>(x_in, x_bf, BSD);
  f32_to_bf16_kernel<<<(unsigned)((FD + 255) / 256), 256, 0, stream>>>(relenc, re_bf, FD);
  zero128_kernel<<<1, 128, 0, stream>>>(lossmx);

  // ---- ln1 over concat(mem, x) ----
  ln_concat_kernel<<<BB * FF, 256, 0, stream>>>(mem_in, x_in, ln1_g, ln1_b, xt_bf, xq_bf);

  // ---- projections ----
  launch_gemm(stream, xt_bf, 0, DD, 0, 0, wt[1], DD, 0, 0, 0,
              nullptr, 0, 0, 0, k_bf, DD, 0, 0, nullptr, BB * FF, DD, DD, 1, 1, 0, 0);
  launch_gemm(stream, xt_bf, 0, DD, 0, 0, wt[3], DD, 0, 0, 0,
              nullptr, 0, 0, 0, v_bf, DD, 0, 0, nullptr, BB * FF, DD, DD, 1, 1, 0, 0);
  launch_gemm(stream, xq_bf, 0, DD, 0, 0, wt[0], DD, 0, 0, 0,
              nullptr, 0, 0, 0, qu_bf, DD, 0, 0, u_param, BB * SS, DD, DD, 1, 1, 0, 0);
  launch_gemm(stream, xq_bf, 0, DD, 0, 0, wt[0], DD, 0, 0, 0,
              nullptr, 0, 0, 0, qv_bf, DD, 0, 0, v_param, BB * SS, DD, DD, 1, 1, 0, 0);
  launch_gemm(stream, re_bf, 0, DD, 0, 0, wt[2], DD, 0, 0, 0,
              nullptr, 0, 0, 0, qrel_bf, DD, 0, 0, nullptr, FF, DD, DD, 1, 1, 0, 0);

  // ---- scores: a_c, then relative-shift scatter-add of b_d_hat ----
  launch_gemm(stream, qu_bf, 0, DD, SD, DHH, k_bf, DD, FD, DHH, 0,
              attn, FF, HSF, SF, nullptr, 0, 0, 0, nullptr,
              SS, FF, DHH, HH, BB * HH, 0, 0);
  launch_gemm(stream, qv_bf, 0, DD, SD, DHH, qrel_bf, DD, 0, DHH, 0,
              attn, FF, HSF, SF, nullptr, 0, 0, 0, nullptr,
              SS, FF, DHH, HH, BB * HH, GF_SHIFT, SS - 1);

  // ---- softmax + attention loss ----
  softmax_kernel<<<BB * HH * SS, 256, 0, stream>>>(attn, lossmx);
  loss_finalize_kernel<<<1, 128, 0, stream>>>(lossmx, loss);

  // ---- ao = attn @ v  (A is f32, converted in staging; B in K x N mode) ----
  launch_gemm(stream, attn, 1, FF, HSF, SF, v_bf, DD, FD, DHH, 1,
              nullptr, 0, 0, 0, ao_bf, DD, SD, DHH, nullptr,
              SS, DHH, FF, HH, BB * HH, 0, 0);

  // ---- rmha_out = gelu(ao @ w_final) ----
  launch_gemm(stream, ao_bf, 0, DD, 0, 0, wt[4], DD, 0, 0, 0,
              nullptr, 0, 0, 0, y_bf, DD, 0, 0, nullptr, BB * SS, DD, DD, 1, 1, GF_GELU, 0);

  // ---- GRU gate 1: g1 = gate(x, y) ----
  launch_gemm(stream, y_bf, 0, DD, 0, 0, wt[7], DD, 0, 0, 0,
              t1, DD, 0, 0, nullptr, 0, 0, 0, nullptr, BB * SS, DD, DD, 1, 1, 0, 0);
  launch_gemm(stream, x_bf, 0, DD, 0, 0, wt[8], DD, 0, 0, 0,
              t1, DD, 0, 0, nullptr, 0, 0, 0, nullptr, BB * SS, DD, DD, 1, 1, GF_ACC, 0);
  launch_gemm(stream, y_bf, 0, DD, 0, 0, wt[9], DD, 0, 0, 0,
              t2, DD, 0, 0, nullptr, 0, 0, 0, nullptr, BB * SS, DD, DD, 1, 1, 0, 0);
  launch_gemm(stream, x_bf, 0, DD, 0, 0, wt[10], DD, 0, 0, 0,
              t2, DD, 0, 0, nullptr, 0, 0, 0, nullptr, BB * SS, DD, DD, 1, 1, GF_ACC, 0);
  gru_rx_kernel<<<(unsigned)((BSD + 255) / 256), 256, 0, stream>>>(t1, x_in, rx_bf, BSD);
  launch_gemm(stream, y_bf, 0, DD, 0, 0, wt[11], DD, 0, 0, 0,
              t3, DD, 0, 0, nullptr, 0, 0, 0, nullptr, BB * SS, DD, DD, 1, 1, 0, 0);
  launch_gemm(stream, rx_bf, 0, DD, 0, 0, wt[12], DD, 0, 0, 0,
              t3, DD, 0, 0, nullptr, 0, 0, 0, nullptr, BB * SS, DD, DD, 1, 1, GF_ACC, 0);
  gru_out_kernel<<<(unsigned)((BSD + 255) / 256), 256, 0, stream>>>(t2, t3, x_in, bg1,
                                                                    g1_f, g1_bf, BSD);

  // ---- MLP: h = gelu(gelu(ln2(g1) @ lin1 + b1) @ lin2 + b2) ----
  ln_simple_kernel<<<BB * SS, 256, 0, stream>>>(g1_f, ln2_g, ln2_b, h0_bf);
  launch_gemm(stream, h0_bf, 0, DD, 0, 0, wt[5], DD, 0, 0, 0,
              nullptr, 0, 0, 0, h1_bf, DD, 0, 0, lin1_b, BB * SS, DD, DD, 1, 1, GF_GELU, 0);
  launch_gemm(stream, h1_bf, 0, DD, 0, 0, wt[6], DD, 0, 0, 0,
              nullptr, 0, 0, 0, h2_bf, DD, 0, 0, lin2_b, BB * SS, DD, DD, 1, 1, GF_GELU, 0);

  // ---- GRU gate 2: out = gate(g1, h) -> d_out ----
  launch_gemm(stream, h2_bf, 0, DD, 0, 0, wt[13], DD, 0, 0, 0,
              t1, DD, 0, 0, nullptr, 0, 0, 0, nullptr, BB * SS, DD, DD, 1, 1, 0, 0);
  launch_gemm(stream, g1_bf, 0, DD, 0, 0, wt[14], DD, 0, 0, 0,
              t1, DD, 0, 0, nullptr, 0, 0, 0, nullptr, BB * SS, DD, DD, 1, 1, GF_ACC, 0);
  launch_gemm(stream, h2_bf, 0, DD, 0, 0, wt[15], DD, 0, 0, 0,
              t2, DD, 0, 0, nullptr, 0, 0, 0, nullptr, BB * SS, DD, DD, 1, 1, 0, 0);
  launch_gemm(stream, g1_bf, 0, DD, 0, 0, wt[16], DD, 0, 0, 0,
              t2, DD, 0, 0, nullptr, 0, 0, 0, nullptr, BB * SS, DD, DD, 1, 1, GF_ACC, 0);
  gru_rx_kernel<<<(unsigned)((BSD + 255) / 256), 256, 0, stream>>>(t1, g1_f, rx_bf, BSD);
  launch_gemm(stream, h2_bf, 0, DD, 0, 0, wt[17], DD, 0, 0, 0,
              t3, DD, 0, 0, nullptr, 0, 0, 0, nullptr, BB * SS, DD, DD, 1, 1, 0, 0);
  launch_gemm(stream, rx_bf, 0, DD, 0, 0, wt[18], DD, 0, 0, 0,
              t3, DD, 0, 0, nullptr, 0, 0, 0, nullptr, BB * SS, DD, DD, 1, 1, GF_ACC, 0);
  gru_out_kernel<<<(unsigned)((BSD + 255) / 256), 256, 0, stream>>>(t2, t3, g1_f, bg2,
                                                                    out_f, nullptr, BSD);
}
